// OptimizedSegmentedExpertFFN_17051020165276
// MI455X (gfx1250) — compile-verified, matrix operand order checked by reading, not run on hardware
//
#include <hip/hip_runtime.h>

#define N_TOKENS 16384
#define D_MODEL  1024
#define N_SEG    4
#define H_SEG    1024

#define TILE_M   64           // tokens per block tile (4 row tiles of 16)
#define CB       128          // output-column block width
#define KC       32           // K chunk (bf16 WMMA K)
#define THREADS  256          // 8 wave32s

#define XS_LD   (D_MODEL + 8) // 1032: row stride 2064B -> 16B aligned, bank-spread
#define HS_LD   (H_SEG + 8)
#define WS_LANE 24            // per-lane stride in swizzled B tile (48B, 16B aligned)
#define NCT     (CB / 16)     // 8 col-tiles per col block
#define WTILE   (NCT * 32 * WS_LANE)

typedef __attribute__((ext_vector_type(16))) __bf16 v16bf;
typedef __attribute__((ext_vector_type(8)))  __bf16 v8bf;
typedef __attribute__((ext_vector_type(8)))  float  v8f;

static __device__ __forceinline__ unsigned short bfbits(float f) {
    unsigned u = __builtin_bit_cast(unsigned, f);
    u += 0x7FFFu + ((u >> 16) & 1u);          // round-to-nearest-even
    return (unsigned short)(u >> 16);
}
static __device__ __forceinline__ __bf16 f2bf(float f) {
    unsigned short h = bfbits(f);
    return __builtin_bit_cast(__bf16, h);
}
static __device__ __forceinline__ unsigned pack2(float a, float b) {
    return (unsigned)bfbits(a) | ((unsigned)bfbits(b) << 16);
}

// A fragment (16x32 bf16, rows striped across lanes): per lane, two contiguous
// 16B runs -> 2x ds_load_b128.
static __device__ __forceinline__ v16bf load_fragA(const __bf16* base, int ld,
                                                   int row0, int k0, int lane) {
    const __bf16* p = base + (size_t)(row0 + (lane & 15)) * ld + k0
                           + ((lane >> 4) << 3);
    v8bf lo = *(const v8bf*)p;
    v8bf hi = *(const v8bf*)(p + 16);
    return __builtin_shufflevector(lo, hi, 0, 1, 2, 3, 4, 5, 6, 7,
                                           8, 9, 10, 11, 12, 13, 14, 15);
}

// B fragment from swizzled LDS weight tile -> 2x ds_load_b128.
static __device__ __forceinline__ v16bf load_fragB(const __bf16* wS,
                                                   int ct, int lane) {
    const __bf16* p = wS + (size_t)(ct * 32 + lane) * WS_LANE;
    v8bf lo = *(const v8bf*)p;
    v8bf hi = *(const v8bf*)(p + 8);
    return __builtin_shufflevector(lo, hi, 0, 1, 2, 3, 4, 5, 6, 7,
                                           8, 9, 10, 11, 12, 13, 14, 15);
}

// Swizzled LDS offset for weight element (k, cc) of a KCxCB tile.
static __device__ __forceinline__ int bswz(int k, int cc) {
    int ct = cc >> 4;
    int L  = (((k >> 3) & 1) << 4) | (cc & 15);
    return (ct * 32 + L) * WS_LANE + ((k >> 4) << 3) + (k & 7);
}

// Stage + swizzle one KCxCB fp32 weight tile into LDS as bf16 (packed stores).
static __device__ __forceinline__ void stage_w(const float* __restrict__ src,
                                               int ld_src, __bf16* dst, int tid) {
#pragma unroll
    for (int idx = tid; idx < (KC / 2) * CB; idx += THREADS) {
        int k  = (idx >> 7) << 1;
        int cc = idx & (CB - 1);
        *(unsigned*)&dst[bswz(k, cc)] =
            pack2(src[(size_t)k * ld_src + cc],
                  src[(size_t)(k + 1) * ld_src + cc]);
    }
}

__global__ void build_lists_kernel(const long long* __restrict__ seg,
                                   int* __restrict__ cnt,
                                   int* __restrict__ list) {
    int i = blockIdx.x * blockDim.x + threadIdx.x;
    if (i < N_TOKENS) {
        int s = (int)seg[i];
        int pos = atomicAdd(&cnt[s], 1);
        list[s * N_TOKENS + pos] = i;
    }
}

__global__ __launch_bounds__(THREADS)
void seg_swiglu_ffn_kernel(const float* __restrict__ x,
                           const float* __restrict__ gate_w,
                           const float* __restrict__ up_w,
                           const float* __restrict__ down_w,
                           const int* __restrict__ cnt,
                           const int* __restrict__ list,
                           float* __restrict__ out) {
    __shared__ alignas(16) __bf16 xs[TILE_M][XS_LD];   // x tile (bf16)     132KB
    __shared__ alignas(16) __bf16 hs[TILE_M][HS_LD];   // SwiGLU hidden     132KB
    __shared__ alignas(16) __bf16 wgS[2][WTILE];       // gate/down tiles    24KB
    __shared__ alignas(16) __bf16 wuS[2][WTILE];       // up tiles           24KB
    __shared__ int tids[TILE_M];

    const int tid  = threadIdx.x;
    const int lane = tid & 31;
    const int wave = tid >> 5;
    const int rb   = wave & 3;              // 4 row tiles of 16
    const int cg   = wave >> 2;             // 2 column groups (4 col-tiles each)

    const int s     = blockIdx.x / (N_TOKENS / TILE_M);
    const int tile  = blockIdx.x % (N_TOKENS / TILE_M);
    const int cnt_s = cnt[s];
    const int tile0 = tile * TILE_M;
    if (tile0 >= cnt_s) return;             // uniform early-exit

    // ---- gather token ids ----
    for (int i = tid; i < TILE_M; i += THREADS) {
        int gr = tile0 + i;
        tids[i] = (gr < cnt_s) ? list[s * N_TOKENS + gr] : -1;
    }
    __syncthreads();

    // ---- stage x tile: float2 global loads, packed 32-bit LDS stores ----
    for (int idx = tid; idx < TILE_M * D_MODEL / 2; idx += THREADS) {
        int r = idx >> 9;
        int c = (idx & (D_MODEL / 2 - 1)) << 1;
        int tok = tids[r];
        float2 v = make_float2(0.f, 0.f);
        if (tok >= 0) v = *(const float2*)(x + (size_t)tok * D_MODEL + c);
        *(unsigned*)&xs[r][c] = pack2(v.x, v.y);
    }

    // ================= Phase 1: hidden = silu(x@Wg) * (x@Wu) =================
    for (int cb = 0; cb < H_SEG / CB; ++cb) {
        const float* gw0 = gate_w + (size_t)s * D_MODEL * H_SEG + cb * CB;
        const float* uw0 = up_w   + (size_t)s * D_MODEL * H_SEG + cb * CB;

        v8f accg[4], accu[4];
#pragma unroll
        for (int t = 0; t < 4; ++t) { accg[t] = {}; accu[t] = {}; }

        __syncthreads();                    // prior consumers of buffers done
        stage_w(gw0, H_SEG, wgS[0], tid);   // prologue: fill buffer 0
        stage_w(uw0, H_SEG, wuS[0], tid);

        const int nk = D_MODEL / KC;
        for (int ki = 0; ki < nk; ++ki) {
            __syncthreads();                // buffer (ki&1) ready, prior reads done
            if (ki + 1 < nk) {              // stage next tile into other buffer
                const float* gw = gw0 + (size_t)(ki + 1) * KC * H_SEG;
                const float* uw = uw0 + (size_t)(ki + 1) * KC * H_SEG;
                __builtin_prefetch(gw + (size_t)KC * H_SEG, 0, 0);
                __builtin_prefetch(uw + (size_t)KC * H_SEG, 0, 0);
                stage_w(gw, H_SEG, wgS[(ki + 1) & 1], tid);
                stage_w(uw, H_SEG, wuS[(ki + 1) & 1], tid);
            }
            const int b = ki & 1;
            v16bf a = load_fragA(&xs[0][0], XS_LD, rb * 16, ki * KC, lane);
#pragma unroll
            for (int t = 0; t < 4; ++t) {
                int ct = cg * 4 + t;
                v16bf bg = load_fragB(wgS[b], ct, lane);
                v16bf bu = load_fragB(wuS[b], ct, lane);
                accg[t] = __builtin_amdgcn_wmma_f32_16x16x32_bf16(
                    false, a, false, bg, (short)0, accg[t], false, false);
                accu[t] = __builtin_amdgcn_wmma_f32_16x16x32_bf16(
                    false, a, false, bu, (short)0, accu[t], false, false);
            }
        }

        // fused SwiGLU epilogue -> hidden tile in LDS (bf16), fast rcp silu
#pragma unroll
        for (int t = 0; t < 4; ++t) {
            int n = cb * CB + (cg * 4 + t) * 16 + (lane & 15);
#pragma unroll
            for (int i = 0; i < 8; ++i) {
                int m = rb * 16 + ((lane >> 4) << 3) + i;
                float g = accg[t][i];
                float u = accu[t][i];
                float h = g * u * __builtin_amdgcn_rcpf(1.0f + __expf(-g));
                hs[m][n] = f2bf(h);
            }
        }
    }

    // ================= Phase 2: out = hidden @ Wd (scatter rows) =============
    for (int cb = 0; cb < D_MODEL / CB; ++cb) {
        const float* dw0 = down_w + (size_t)s * H_SEG * D_MODEL + cb * CB;

        v8f acc[4];
#pragma unroll
        for (int t = 0; t < 4; ++t) acc[t] = {};

        __syncthreads();
        stage_w(dw0, D_MODEL, wgS[0], tid);

        const int nk = H_SEG / KC;
        for (int ki = 0; ki < nk; ++ki) {
            __syncthreads();
            if (ki + 1 < nk) {
                const float* dw = dw0 + (size_t)(ki + 1) * KC * D_MODEL;
                __builtin_prefetch(dw + (size_t)KC * D_MODEL, 0, 0);
                stage_w(dw, D_MODEL, wgS[(ki + 1) & 1], tid);
            }
            const int b = ki & 1;
            v16bf a = load_fragA(&hs[0][0], HS_LD, rb * 16, ki * KC, lane);
#pragma unroll
            for (int t = 0; t < 4; ++t) {
                v16bf bd = load_fragB(wgS[b], cg * 4 + t, lane);
                acc[t] = __builtin_amdgcn_wmma_f32_16x16x32_bf16(
                    false, a, false, bd, (short)0, acc[t], false, false);
            }
        }

        // scatter f32 results to assigned token rows
#pragma unroll
        for (int t = 0; t < 4; ++t) {
            int n = cb * CB + (cg * 4 + t) * 16 + (lane & 15);
#pragma unroll
            for (int i = 0; i < 8; ++i) {
                int m = rb * 16 + ((lane >> 4) << 3) + i;
                int tok = tids[m];
                if (tok >= 0) out[(size_t)tok * D_MODEL + n] = acc[t][i];
            }
        }
    }
}

extern "C" void kernel_launch(void* const* d_in, const int* in_sizes, int n_in,
                              void* d_out, int out_size, void* d_ws, size_t ws_size,
                              hipStream_t stream) {
    const float*     x      = (const float*)d_in[0];
    const float*     gate_w = (const float*)d_in[1];
    const float*     up_w   = (const float*)d_in[2];
    const float*     down_w = (const float*)d_in[3];
    const long long* seg    = (const long long*)d_in[4];
    float*           out    = (float*)d_out;

    int* cnt  = (int*)d_ws;        // 8 ints (4 used, padded)
    int* list = cnt + 8;           // N_SEG * N_TOKENS ints

    hipMemsetAsync(d_ws, 0, 8 * sizeof(int), stream);
    build_lists_kernel<<<(N_TOKENS + 255) / 256, 256, 0, stream>>>(seg, cnt, list);

    dim3 grid(N_SEG * (N_TOKENS / TILE_M));
    seg_swiglu_ffn_kernel<<<grid, THREADS, 0, stream>>>(
        x, gate_w, up_w, down_w, cnt, list, out);
}
